// ScaledDotAttention_29274497090056
// MI455X (gfx1250) — compile-verified, matrix operand order checked by reading, not run on hardware
//
#include <hip/hip_runtime.h>
#include <hip/hip_bf16.h>

typedef __attribute__((ext_vector_type(16))) _Float16 v16h;
typedef __attribute__((ext_vector_type(8)))  _Float16 v8h;
typedef __attribute__((ext_vector_type(8)))  float    v8f;
typedef __attribute__((ext_vector_type(4)))  unsigned int u32x4;
typedef __attribute__((ext_vector_type(8)))  int      i32x8;
typedef __attribute__((ext_vector_type(4)))  int      i32x4;

#define SK 8192
#define SQ 8192
#define DD 128
// 1/sqrt(128) * log2(e): softmax computed in log2 domain -> native v_exp_f32
#define SCALE2 0.12751589906834162f

#if defined(__HIP_DEVICE_COMPILE__) && __has_builtin(__builtin_amdgcn_tensor_load_to_lds) && \
    __has_builtin(__builtin_amdgcn_s_wait_tensorcnt)
#define USE_TDM 1
#else
#define USE_TDM 0
#endif

// Raw v_exp_f32: args here are bounded (<= 0, > -40), so skip OCML's
// underflow-guard path (saves ~6 VALU per exponential in the hot loop).
__device__ __forceinline__ float fexp2(float x) {
#if defined(__HIP_DEVICE_COMPILE__) && __has_builtin(__builtin_amdgcn_exp2f)
  return __builtin_amdgcn_exp2f(x);
#else
  return exp2f(x);
#endif
}

// ---------------------------------------------------------------------------
// Load a 16x32 f16 tile from a row-major matrix into the WMMA A/B register
// layout (ISA 7.12.2, 16-bit A-matrix 16x32):
//   lanes 0-15 : row = row0+lane,    halves 0-7 = cols c0+0..7,  8-15 = c0+16..23
//   lanes 16-31: row = row0+lane-16, halves 0-7 = cols c0+8..15, 8-15 = c0+24..31
// Both chunks are 16B aligned -> two b128 loads per lane (global or ds).
// Serves B operands too: a B (KxN) tile equals the A layout of rows of the
// row-major N-major source (lane = column index n).
// ---------------------------------------------------------------------------
__device__ __forceinline__ v16h load_tile_ab(const _Float16* base,
                                             int ld, int row0, int col0, int lane) {
  const int r  = row0 + (lane & 15);
  const int hs = (lane >> 4) & 1;
  const _Float16* p = base + (size_t)r * ld + col0 + hs * 8;
  v8h lo = *(const v8h*)p;
  v8h hi = *(const v8h*)(p + 16);
  v16h out;
#pragma unroll
  for (int i = 0; i < 8; ++i) { out[i] = lo[i]; out[i + 8] = hi[i]; }
  return out;
}

#if USE_TDM
// Issue a TDM load of a 2D tile (tile_dim0=32 halfs wide, tile_dim1=128 rows)
// from row-major Vt[128][8192] (f16) into LDS at lds_byte_addr.
// D# bitfields per CDNA5 ISA 8.3/8.4 (group0: count/lds/global/type,
// group1: data_size, tensor dims, tile dims, dim0 stride).
// This toolchain's builtin takes 6 args: (g0 u32x4, g1 i32x8, i32x4, i32x4,
// i32x8, cpol) — trailing groups zero for a 2-D tensor.
__device__ __forceinline__ void tdm_load_vt(const _Float16* gsrc, unsigned int lds_byte_addr) {
  unsigned long long ga = (unsigned long long)(uintptr_t)gsrc;
  u32x4 g0;
  g0[0] = 1u;                                            // count=1, user mode
  g0[1] = lds_byte_addr;                                 // lds_addr
  g0[2] = (unsigned int)ga;                              // global_addr[31:0]
  g0[3] = (unsigned int)((ga >> 32) & 0x01FFFFFFu) | (2u << 30);  // addr[56:32] | type=2
  i32x8 g1;
  g1[0] = (int)(1u << 16);                               // data_size=1 (2 bytes)
  g1[1] = (int)(((unsigned)SQ & 0xFFFFu) << 16);         // tensor_dim0 lo16
  g1[2] = (int)(((unsigned)SQ >> 16) | (((unsigned)DD & 0xFFFFu) << 16)); // dim0 hi | dim1 lo
  g1[3] = (int)(((unsigned)DD >> 16) | (32u << 16));     // dim1 hi | tile_dim0=32
  g1[4] = (int)128;                                      // tile_dim1=128, tile_dim2=0
  g1[5] = (int)SQ;                                       // tensor_dim0_stride lo32
  g1[6] = 0;                                             // stride hi | dim1_stride lo
  g1[7] = 0;
  i32x4 z4 = {0, 0, 0, 0};
  i32x8 z8 = {0, 0, 0, 0, 0, 0, 0, 0};
  __builtin_amdgcn_tensor_load_to_lds(g0, g1, z4, z4, z8, 0);
}
#endif

// ---------------------------------------------------------------------------
// Kernel 1: f32 -> f16 copies of K and Q; transposed f16 copy of V (Vt[d][q]).
// ---------------------------------------------------------------------------
__global__ void sda_convert(const float* __restrict__ K, const float* __restrict__ Q,
                            const float* __restrict__ V,
                            _Float16* __restrict__ Kh, _Float16* __restrict__ Qh,
                            _Float16* __restrict__ Vt) {
  int i = blockIdx.x * blockDim.x + threadIdx.x;   // 0 .. SK*DD-1
  Kh[i] = (_Float16)K[i];
  Qh[i] = (_Float16)Q[i];
  int q = i >> 7, d = i & 127;
  Vt[(size_t)d * SQ + q] = (_Float16)V[i];
}

// ---------------------------------------------------------------------------
// Kernel 2: per-query-column softmax stats over the k axis.
// Each wave owns 16 q columns; loops all k in 16-row tiles.
// Online (max, sum-exp2) per column, merged across lane halves via shfl_xor 16.
// ---------------------------------------------------------------------------
__global__ void sda_stats(const _Float16* __restrict__ Kh,
                          const _Float16* __restrict__ Qh,
                          float* __restrict__ m2, float* __restrict__ rZ) {
  const int lane = threadIdx.x & 31;
  const int wave = (blockIdx.x * blockDim.x + threadIdx.x) >> 5;
  const int q0 = wave * 16;

  v16h bq[4];
#pragma unroll
  for (int dc = 0; dc < 4; ++dc) bq[dc] = load_tile_ab(Qh, DD, q0, dc * 32, lane);

  float mrun = -__builtin_inff();
  float zrun = 0.0f;

  for (int k0 = 0; k0 < SK; k0 += 16) {
    v8f c = {};
#pragma unroll
    for (int dc = 0; dc < 4; ++dc) {
      v16h a = load_tile_ab(Kh, DD, k0, dc * 32, lane);
      c = __builtin_amdgcn_wmma_f32_16x16x32_f16(false, a, false, bq[dc],
                                                 (short)0, c, false, false);
    }
    // C layout: lane holds column n=lane&15, rows r+8*(lane>>4), r=0..7
    float tmax = c[0];
#pragma unroll
    for (int r = 1; r < 8; ++r) tmax = fmaxf(tmax, c[r]);
    tmax = fmaxf(tmax, __shfl_xor(tmax, 16, 32));
    tmax *= SCALE2;
    float nm = fmaxf(mrun, tmax);
    float ts = 0.0f;
#pragma unroll
    for (int r = 0; r < 8; ++r) ts += fexp2(c[r] * SCALE2 - nm);
    ts += __shfl_xor(ts, 16, 32);
    zrun = zrun * fexp2(mrun - nm) + ts;
    mrun = nm;
  }
  if (lane < 16) {
    m2[q0 + lane] = mrun;
    rZ[q0 + lane] = 1.0f / zrun;
  }
}

// ---------------------------------------------------------------------------
// Kernel 3: output. Each wave owns 16 k rows x full 128 d (8 f32 accumulators).
// Per 32-q chunk: recompute S, p = exp2(s-m2[q])*rZ[q] in f16,
// C-layout -> A-layout via wave-private LDS tile, then 8 WMMAs against Vt.
// Vt chunk (128x32 f16, shared by ALL waves) is staged block-wide into LDS by
// the Tensor Data Mover, double-buffered to overlap with the S GEMM.
// ---------------------------------------------------------------------------
__global__ void sda_output(const _Float16* __restrict__ Kh,
                           const _Float16* __restrict__ Qh,
                           const _Float16* __restrict__ Vt,
                           const float* __restrict__ m2,
                           const float* __restrict__ rZ,
                           float* __restrict__ out) {
  __shared__ _Float16 plds[8][16 * 32];       // per-wave 16x32 P tile, 8 KB
#if USE_TDM
  __shared__ _Float16 vstage[2][DD * 32];     // double-buffered Vt chunk, 16 KB
#endif
  const int lane = threadIdx.x & 31;
  const int wv   = threadIdx.x >> 5;
  const int wave = (blockIdx.x * blockDim.x + threadIdx.x) >> 5;
  const int k0   = wave * 16;
  const int n    = lane & 15;
  const int hi   = lane >> 4;

  v16h ak[4];
#pragma unroll
  for (int dc = 0; dc < 4; ++dc) ak[dc] = load_tile_ab(Kh, DD, k0, dc * 32, lane);

  v8f acc[8];
#pragma unroll
  for (int dt = 0; dt < 8; ++dt) acc[dt] = (v8f){};

#if USE_TDM
  if (wv == 0) tdm_load_vt(Vt, (unsigned int)(uintptr_t)&vstage[0][0]);
#endif

  const int NCHUNK = SQ / 32;
  for (int ci = 0; ci < NCHUNK; ++ci) {
    const int q0 = ci * 32;
    // ---- S = K x Q^T for 16 k-rows x 32 q-cols (accumulate over d=128) ----
    v8f c0 = {}, c1 = {};
#pragma unroll
    for (int dc = 0; dc < 4; ++dc) {
      v16h b0 = load_tile_ab(Qh, DD, q0,      dc * 32, lane);
      v16h b1 = load_tile_ab(Qh, DD, q0 + 16, dc * 32, lane);
      c0 = __builtin_amdgcn_wmma_f32_16x16x32_f16(false, ak[dc], false, b0,
                                                  (short)0, c0, false, false);
      c1 = __builtin_amdgcn_wmma_f32_16x16x32_f16(false, ak[dc], false, b1,
                                                  (short)0, c1, false, false);
    }
    __syncthreads();   // (A) prior chunk's plds/vstage reads complete
    // ---- p = exp2(s*scale2 - m2[q]) * rZ[q], staged to LDS in (m,q) form --
    float mq0 = m2[q0 + n],      rz0 = rZ[q0 + n];
    float mq1 = m2[q0 + 16 + n], rz1 = rZ[q0 + 16 + n];
#pragma unroll
    for (int r = 0; r < 8; ++r) {
      int mrow = r + 8 * hi;
      plds[wv][mrow * 32 + n]      = (_Float16)(fexp2(c0[r] * SCALE2 - mq0) * rz0);
      plds[wv][mrow * 32 + 16 + n] = (_Float16)(fexp2(c1[r] * SCALE2 - mq1) * rz1);
    }
#if USE_TDM
    if (wv == 0) __builtin_amdgcn_s_wait_tensorcnt(0);  // chunk ci resident
#endif
    __syncthreads();   // (B) plds written + vstage[ci&1] published to block
#if USE_TDM
    if (wv == 0 && ci + 1 < NCHUNK)                     // prefetch next chunk
      tdm_load_vt(Vt + (size_t)(q0 + 32), (unsigned int)(uintptr_t)&vstage[(ci + 1) & 1][0]);
#endif
    // ---- read P back in WMMA A layout (16x32) ----------------------------
    v16h pa;
    {
      v8h lo = *(const v8h*)&plds[wv][n * 32 + 8 * hi];
      v8h hp = *(const v8h*)&plds[wv][n * 32 + 16 + 8 * hi];
#pragma unroll
      for (int i = 0; i < 8; ++i) { pa[i] = lo[i]; pa[i + 8] = hp[i]; }
    }
    // ---- O(16x128) += P(16x32) x Vt-as-B(32q x 16d) per d-tile -----------
#if USE_TDM
    const _Float16* vsrc = &vstage[ci & 1][0];          // row-major [128][32] in LDS
#pragma unroll
    for (int dt = 0; dt < 8; ++dt) {
      v16h bv = load_tile_ab(vsrc, 32, dt * 16, 0, lane);
      acc[dt] = __builtin_amdgcn_wmma_f32_16x16x32_f16(false, pa, false, bv,
                                                       (short)0, acc[dt], false, false);
    }
#else
#pragma unroll
    for (int dt = 0; dt < 8; ++dt) {
      v16h bv = load_tile_ab(Vt, SQ, dt * 16, q0, lane);
      acc[dt] = __builtin_amdgcn_wmma_f32_16x16x32_f16(false, pa, false, bv,
                                                       (short)0, acc[dt], false, false);
    }
#endif
  }
  // ---- store C layout: element (m = r+8*hi, col = dt*16+n); write-once ----
#pragma unroll
  for (int dt = 0; dt < 8; ++dt) {
#pragma unroll
    for (int r = 0; r < 8; ++r) {
      __builtin_nontemporal_store(acc[dt][r],
          &out[(size_t)(k0 + r + 8 * hi) * DD + dt * 16 + n]);
    }
  }
}

extern "C" void kernel_launch(void* const* d_in, const int* in_sizes, int n_in,
                              void* d_out, int out_size, void* d_ws, size_t ws_size,
                              hipStream_t stream) {
  (void)in_sizes; (void)n_in; (void)out_size; (void)ws_size;
  const float* K = (const float*)d_in[0];
  const float* Q = (const float*)d_in[1];
  const float* V = (const float*)d_in[2];
  float* out = (float*)d_out;

  _Float16* Kh = (_Float16*)d_ws;                       // [8192,128] f16
  _Float16* Qh = Kh + (size_t)SK * DD;                  // [8192,128] f16
  _Float16* Vt = Qh + (size_t)SQ * DD;                  // [128,8192] f16 (transposed)
  float*    m2 = (float*)(Vt + (size_t)DD * SQ);        // [8192] column max (log2 dom)
  float*    rZ = m2 + SQ;                               // [8192] 1/sum

  sda_convert<<<(SK * DD) / 256, 256, 0, stream>>>(K, Q, V, Kh, Qh, Vt);
  // 512 waves each: 8 waves/block -> 64 blocks
  sda_stats<<<64, 256, 0, stream>>>(Kh, Qh, m2, rZ);
  sda_output<<<64, 256, 0, stream>>>(Kh, Qh, Vt, m2, rZ, out);
}
